// GAT_Arxiv_46076409152400
// MI455X (gfx1250) — compile-verified, matrix operand order checked by reading, not run on hardware
//
#include <hip/hip_runtime.h>
#include <hip/hip_bf16.h>
#include <stdint.h>

#define N_NODES 50000
#define N_EDGES 800000
#define ET_EDGES (N_EDGES + N_NODES)   // with self loops
#define NEG_SLOPE 0.2f
#define BN_EPS 1e-5f

typedef __attribute__((ext_vector_type(16))) __bf16   bf16x16;
typedef __attribute__((ext_vector_type(8)))  float    f32x8;
typedef __attribute__((ext_vector_type(8)))  uint32_t u32x8;

// ---------------- helpers ----------------

__device__ __forceinline__ unsigned short f2bf(float f) {
  uint32_t x = __float_as_uint(f);
  uint32_t r = x + 0x7FFFu + ((x >> 16) & 1u);      // round-to-nearest-even
  if ((x & 0x7F800000u) == 0x7F800000u) r = x;       // inf/nan passthrough
  return (unsigned short)(r >> 16);
}

__device__ __forceinline__ void atomicMaxF(float* addr, float val) {
  // IEEE ordering trick: int max for >=0, uint min for <0 (init must be -inf)
  if (val >= 0.0f) atomicMax((int*)addr, __float_as_int(val));
  else             atomicMin((unsigned int*)addr, __float_as_uint(val));
}

// async global->LDS copy of 16 bytes (per lane), tracked with ASYNCcnt
__device__ __forceinline__ void async_copy_b128(uint32_t lds_off, const void* gptr) {
  uint64_t ga = (uint64_t)(uintptr_t)gptr;
  asm volatile("global_load_async_to_lds_b128 %0, %1, off"
               :: "v"(lds_off), "v"(ga) : "memory");
}
__device__ __forceinline__ void async_wait0() {
  asm volatile("s_wait_asynccnt 0x0" ::: "memory");
}
__device__ __forceinline__ uint32_t lds_off32(const void* p) {
  // generic shared address: LDS aperture calc uses addr[31:0] as LDS offset
  return (uint32_t)(uintptr_t)p;
}

// ---------------- elementwise / init kernels ----------------

__global__ void k_fill_f32(float* p, float v, int n) {
  int i = blockIdx.x * blockDim.x + threadIdx.x;
  int stride = gridDim.x * blockDim.x;
  for (; i < n; i += stride) p[i] = v;
}

__global__ void k_bias_init(float* out, const float* __restrict__ bias, int nn, int d) {
  int i = blockIdx.x * blockDim.x + threadIdx.x;
  int stride = gridDim.x * blockDim.x;
  int tot = nn * d;
  for (; i < tot; i += stride) out[i] = bias[i % d];
}

__global__ void k_cvt_bf16(const float* __restrict__ in, unsigned short* __restrict__ out, int n) {
  int i = blockIdx.x * blockDim.x + threadIdx.x;
  int stride = gridDim.x * blockDim.x;
  for (; i < n; i += stride) out[i] = f2bf(in[i]);
}

// W [K x Nc] row-major  ->  Wt [Nc x K] bf16
__global__ void k_transpose_cvt(const float* __restrict__ W, unsigned short* __restrict__ Wt,
                                int K, int Nc) {
  int i = blockIdx.x * blockDim.x + threadIdx.x;
  int stride = gridDim.x * blockDim.x;
  int tot = K * Nc;
  for (; i < tot; i += stride) {
    int k = i / Nc, n = i % Nc;
    Wt[(size_t)n * K + k] = f2bf(W[i]);
  }
}

// ---------------- LDS-tiled WMMA GEMM ----------------
// C[M,Nc] = A[M,K] * B[K,Nc].  A bf16 row-major (u32 word pairs), Bt bf16 [Nc x K].
// Block = 256 threads = 8 waves; macro tile 128(M) x 128(N); wave grid 2x4;
// each wave: 4(M) x 2(N) WMMA tiles, A fragment reused across N.
#define BM 128
#define BN 128
#define BK 32          // K elements per stage
#define BK_W 16        // u32 words per row per stage

__global__ __launch_bounds__(256) void k_wmma_gemm_tiled(
    const uint32_t* __restrict__ A, const uint32_t* __restrict__ Bt,
    float* __restrict__ C, int M, int K, int Nc) {
  __shared__ __align__(16) uint32_t As[BM * BK_W];   // 8 KB
  __shared__ __align__(16) uint32_t Bs[BN * BK_W];   // 8 KB

  int K2   = K >> 1;
  int bm   = blockIdx.x;
  int bn   = blockIdx.y;
  int tid  = threadIdx.x;
  int wave = tid >> 5, lane = tid & 31;
  int wm   = wave >> 2;       // 0..1 -> M offset wm*64
  int wn   = wave & 3;        // 0..3 -> N offset wn*32

  f32x8 acc[4][2] = {};

  // stage-copy mapping: thread covers row = tid>>1, word-half = (tid&1)*8
  int crow  = tid >> 1;
  int chalf = (tid & 1) * 8;
  int aRowG = bm * BM + crow; if (aRowG >= M)  aRowG = 0;   // clamp, masked on store
  int bRowG = bn * BN + crow; if (bRowG >= Nc) bRowG = 0;
  const uint32_t* aSrc = A  + (size_t)aRowG * K2 + chalf;
  const uint32_t* bSrc = Bt + (size_t)bRowG * K2 + chalf;
  uint32_t aDst = lds_off32(&As[crow * BK_W + chalf]);
  uint32_t bDst = lds_off32(&Bs[crow * BK_W + chalf]);

  int kw0 = (lane >= 16) ? 4 : 0;   // word offset of this lane-half's K group
  int l15 = lane & 15;

  for (int kb = 0; kb < K; kb += BK) {
    // async-stage 16 KB (A+B) into LDS: 4x b128 per thread
    async_copy_b128(aDst,       aSrc);
    async_copy_b128(aDst + 16u, aSrc + 4);
    async_copy_b128(bDst,       bSrc);
    async_copy_b128(bDst + 16u, bSrc + 4);
    async_wait0();
    __syncthreads();
    aSrc += BK_W; bSrc += BK_W;

    u32x8 bfrag[2];
#pragma unroll
    for (int j = 0; j < 2; ++j) {
      const uint32_t* bp = &Bs[(wn * 32 + j * 16 + l15) * BK_W + kw0];
      uint4 lo = *(const uint4*)bp;
      uint4 hi = *(const uint4*)(bp + 8);
      u32x8 f = {lo.x, lo.y, lo.z, lo.w, hi.x, hi.y, hi.z, hi.w};
      bfrag[j] = f;
    }
#pragma unroll
    for (int i = 0; i < 4; ++i) {
      const uint32_t* ap = &As[(wm * 64 + i * 16 + l15) * BK_W + kw0];
      uint4 lo = *(const uint4*)ap;
      uint4 hi = *(const uint4*)(ap + 8);
      u32x8 afrag = {lo.x, lo.y, lo.z, lo.w, hi.x, hi.y, hi.z, hi.w};
#pragma unroll
      for (int j = 0; j < 2; ++j)
        acc[i][j] = __builtin_amdgcn_wmma_f32_16x16x32_bf16(
            false, __builtin_bit_cast(bf16x16, afrag),
            false, __builtin_bit_cast(bf16x16, bfrag[j]),
            (short)0, acc[i][j], false, false);
    }
    __syncthreads();
  }

  // C/D layout: VGPR r -> (M=r | r+8, N=lane&15) by lane half
  int mh = (lane >> 4) ? 8 : 0;
#pragma unroll
  for (int i = 0; i < 4; ++i) {
    int m0 = bm * BM + wm * 64 + i * 16 + mh;
#pragma unroll
    for (int j = 0; j < 2; ++j) {
      int n_out = bn * BN + wn * 32 + j * 16 + l15;
      if (n_out < Nc) {
#pragma unroll
        for (int r = 0; r < 8; ++r) {
          int m = m0 + r;
          if (m < M) C[(size_t)m * Nc + n_out] = acc[i][j][r];
        }
      }
    }
  }
}

// ---------------- attention coefficients ----------------
__global__ void k_attn_coef(const float* __restrict__ h, const float* __restrict__ a_src,
                            const float* __restrict__ a_dst, float* __restrict__ as_n,
                            float* __restrict__ ad_n, int nn, int H, int C) {
  int i = blockIdx.x * blockDim.x + threadIdx.x;
  if (i >= nn * H) return;
  int n = i / H, hd = i % H;
  const float* row = h + (size_t)n * H * C + (size_t)hd * C;
  const float* avs = a_src + hd * C;
  const float* avd = a_dst + hd * C;
  float s = 0.f, d = 0.f;
  for (int c = 0; c < C; ++c) { float v = row[c]; s += v * avs[c]; d += v * avd[c]; }
  as_n[i] = s; ad_n[i] = d;
}

// ---------------- edge passes ----------------
__device__ __forceinline__ int edge_src(const long long* sa, int e) {
  return (e < N_EDGES) ? (int)sa[e] : (e - N_EDGES);
}
__device__ __forceinline__ int edge_dst(const long long* da, int e) {
  return (e < N_EDGES) ? (int)da[e] : (e - N_EDGES);
}

__global__ void k_edge_logits(const long long* __restrict__ sa, const long long* __restrict__ da,
                              const float* __restrict__ as_n, const float* __restrict__ ad_n,
                              float* __restrict__ ebuf, float* __restrict__ emax, int H) {
  int i = blockIdx.x * blockDim.x + threadIdx.x;
  int stride = gridDim.x * blockDim.x;
  int tot = ET_EDGES * H;
  for (; i < tot; i += stride) {
    int e = i / H, hd = i % H;
    int s = edge_src(sa, e), d = edge_dst(da, e);
    float v = as_n[s * H + hd] + ad_n[d * H + hd];
    v = (v > 0.f) ? v : NEG_SLOPE * v;
    ebuf[i] = v;
    atomicMaxF(&emax[d * H + hd], v);
  }
}

__global__ void k_edge_exp(const long long* __restrict__ sa, const long long* __restrict__ da,
                           float* __restrict__ ebuf, const float* __restrict__ emax,
                           float* __restrict__ denom, int H) {
  int i = blockIdx.x * blockDim.x + threadIdx.x;
  int stride = gridDim.x * blockDim.x;
  int tot = ET_EDGES * H;
  for (; i < tot; i += stride) {
    int e = i / H, hd = i % H;
    int d = edge_dst(da, e);
    float v = __expf(ebuf[i] - emax[d * H + hd]);
    ebuf[i] = v;
    atomicAdd(&denom[d * H + hd], v);
  }
}

// out[dst] += (e/denom[dst]) * h[src]; one wave per edge, head-major feature loop
__global__ void k_edge_aggregate(const long long* __restrict__ sa, const long long* __restrict__ da,
                                 const float* __restrict__ ebuf, const float* __restrict__ denom,
                                 const float* __restrict__ h, float* __restrict__ out,
                                 int H, int C) {
  int dfeat  = H * C;
  int wave   = (int)((blockIdx.x * blockDim.x + threadIdx.x) >> 5);
  int lane   = threadIdx.x & 31;
  int nwaves = (int)((gridDim.x * blockDim.x) >> 5);
  for (int e = wave; e < ET_EDGES; e += nwaves) {
    int s = edge_src(sa, e), d = edge_dst(da, e);
    const float* hrow = h + (size_t)s * dfeat;
    float* orow = out + (size_t)d * dfeat;
    for (int hd = 0; hd < H; ++hd) {
      float alpha = ebuf[(size_t)e * H + hd] / (denom[d * H + hd] + 1e-16f);
      int base = hd * C;
      for (int f = lane; f < C; f += 32)
        atomicAdd(&orow[base + f], alpha * hrow[base + f]);
    }
  }
}

// ---------------- BN (eval) + ELU, in place ----------------
__global__ void k_bn_elu(float* __restrict__ x, const float* __restrict__ g,
                         const float* __restrict__ bt, const float* __restrict__ m,
                         const float* __restrict__ v, int nn, int d) {
  int i = blockIdx.x * blockDim.x + threadIdx.x;
  int stride = gridDim.x * blockDim.x;
  int tot = nn * d;
  for (; i < tot; i += stride) {
    int f = i % d;
    float y = (x[i] - m[f]) * g[f] * rsqrtf(v[f] + BN_EPS) + bt[f];
    x[i] = (y > 0.f) ? y : (__expf(y) - 1.f);
  }
}

// ---------------- head mean + bias + log_softmax ----------------
__global__ void k_head_mean_lsm(const float* __restrict__ agg, const float* __restrict__ b2,
                                float* __restrict__ out, int nn) {
  int n = blockIdx.x * blockDim.x + threadIdx.x;
  if (n >= nn) return;
  float logits[40];
#pragma unroll
  for (int c = 0; c < 40; ++c) {
    float s = 0.f;
    for (int hd = 0; hd < 6; ++hd) s += agg[(size_t)n * 240 + hd * 40 + c];
    logits[c] = s * (1.f / 6.f) + b2[c];
  }
  float mx = -INFINITY;
  for (int c = 0; c < 40; ++c) mx = fmaxf(mx, logits[c]);
  float se = 0.f;
  for (int c = 0; c < 40; ++c) se += __expf(logits[c] - mx);
  float lse = mx + __logf(se);
  for (int c = 0; c < 40; ++c) out[(size_t)n * 40 + c] = logits[c] - lse;
}

// ---------------- host orchestration ----------------

static inline int cdiv(int a, int b) { return (a + b - 1) / b; }

struct LayerBufs {
  unsigned short* xb;   // bf16 input       [N x K]
  unsigned short* wt;   // bf16 W^T         [Nc x K]
  float* h;             // gemm out         [N x H*C]
  float* agg;           // aggregated out   [N x H*C]
  float* as_n; float* ad_n; float* emax; float* denom; float* ebuf;
};

static void run_gat_layer(const float* x_in, int K, const float* W,
                          const float* a_s, const float* a_d, const float* bias,
                          int H, int C, bool with_bias,
                          const long long* sa, const long long* da,
                          const LayerBufs& B, hipStream_t st) {
  int d = H * C;
  // 1. convert input + transpose-convert weights to bf16
  k_cvt_bf16<<<cdiv(N_NODES * K, 256), 256, 0, st>>>(x_in, B.xb, N_NODES * K);
  k_transpose_cvt<<<cdiv(K * d, 256), 256, 0, st>>>(W, B.wt, K, d);
  // 2. LDS-tiled WMMA GEMM: h = x @ W
  dim3 grid(cdiv(N_NODES, BM), cdiv(d, BN));
  k_wmma_gemm_tiled<<<grid, 256, 0, st>>>((const uint32_t*)B.xb, (const uint32_t*)B.wt,
                                          B.h, N_NODES, K, d);
  // 3. per-node attention coefficients
  k_attn_coef<<<cdiv(N_NODES * H, 256), 256, 0, st>>>(B.h, a_s, a_d, B.as_n, B.ad_n,
                                                      N_NODES, H, C);
  // 4. init segment buffers + output accumulator
  k_fill_f32<<<cdiv(N_NODES * H, 256), 256, 0, st>>>(B.emax, -INFINITY, N_NODES * H);
  k_fill_f32<<<cdiv(N_NODES * H, 256), 256, 0, st>>>(B.denom, 0.f, N_NODES * H);
  if (with_bias)
    k_bias_init<<<4096, 256, 0, st>>>(B.agg, bias, N_NODES, d);
  else
    k_fill_f32<<<4096, 256, 0, st>>>(B.agg, 0.f, N_NODES * d);
  // 5. edge passes (segment softmax + weighted aggregation)
  k_edge_logits<<<4096, 256, 0, st>>>(sa, da, B.as_n, B.ad_n, B.ebuf, B.emax, H);
  k_edge_exp<<<4096, 256, 0, st>>>(sa, da, B.ebuf, B.emax, B.denom, H);
  k_edge_aggregate<<<8192, 256, 0, st>>>(sa, da, B.ebuf, B.denom, B.h, B.agg, H, C);
}

extern "C" void kernel_launch(void* const* d_in, const int* in_sizes, int n_in,
                              void* d_out, int out_size, void* d_ws, size_t ws_size,
                              hipStream_t stream) {
  const float*     x  = (const float*)d_in[0];
  const long long* ei = (const long long*)d_in[1];
  const float *W0 = (const float*)d_in[2],  *as0 = (const float*)d_in[3],
              *ad0 = (const float*)d_in[4], *b0  = (const float*)d_in[5],
              *g0  = (const float*)d_in[6], *bt0 = (const float*)d_in[7],
              *m0  = (const float*)d_in[8], *v0  = (const float*)d_in[9],
              *W1  = (const float*)d_in[10],*as1 = (const float*)d_in[11],
              *ad1 = (const float*)d_in[12],*b1  = (const float*)d_in[13],
              *g1  = (const float*)d_in[14],*bt1 = (const float*)d_in[15],
              *m1  = (const float*)d_in[16],*v1  = (const float*)d_in[17],
              *W2  = (const float*)d_in[18],*as2 = (const float*)d_in[19],
              *ad2 = (const float*)d_in[20],*b2  = (const float*)d_in[21];

  const long long* sa = ei;            // src row
  const long long* da = ei + N_EDGES;  // dst row

  // carve workspace (256B aligned slabs)
  char* w = (char*)d_ws;
  auto carve = [&](size_t bytes) { char* p = w; w += (bytes + 255) & ~(size_t)255; return p; };
  float* h_buf   = (float*)carve((size_t)N_NODES * 512 * 4);
  float* agg_buf = (float*)carve((size_t)N_NODES * 512 * 4);
  unsigned short* xb = (unsigned short*)carve((size_t)N_NODES * 512 * 2);
  unsigned short* wt = (unsigned short*)carve((size_t)512 * 512 * 2);
  float* as_n  = (float*)carve((size_t)N_NODES * 6 * 4);
  float* ad_n  = (float*)carve((size_t)N_NODES * 6 * 4);
  float* emax  = (float*)carve((size_t)N_NODES * 6 * 4);
  float* denom = (float*)carve((size_t)N_NODES * 6 * 4);
  float* ebuf  = (float*)carve((size_t)ET_EDGES * 6 * 4);

  LayerBufs B{xb, wt, h_buf, agg_buf, as_n, ad_n, emax, denom, ebuf};

  // ----- layer 0: GATConv(128 -> 128, H=4, concat) + BN + ELU -----
  run_gat_layer(x, 128, W0, as0, ad0, b0, 4, 128, true, sa, da, B, stream);
  k_bn_elu<<<4096, 256, 0, stream>>>(agg_buf, g0, bt0, m0, v0, N_NODES, 512);

  // ----- layer 1: GATConv(512 -> 128, H=4, concat) + BN + ELU -----
  run_gat_layer(agg_buf, 512, W1, as1, ad1, b1, 4, 128, true, sa, da, B, stream);
  k_bn_elu<<<4096, 256, 0, stream>>>(agg_buf, g1, bt1, m1, v1, N_NODES, 512);

  // ----- layer 2: GATConv(512 -> 40, H=6, concat=False) -----
  run_gat_layer(agg_buf, 512, W2, as2, ad2, b2, 6, 40, false, sa, da, B, stream);

  // ----- head mean + bias + log_softmax -----
  k_head_mean_lsm<<<cdiv(N_NODES, 128), 128, 0, stream>>>(agg_buf, b2, (float*)d_out, N_NODES);
}
